// ContextualAttentionLayer_51539607552749
// MI455X (gfx1250) — compile-verified, matrix operand order checked by reading, not run on hardware
//
#include <hip/hip_runtime.h>
#include <hip/hip_bf16.h>

typedef __bf16 v16bf __attribute__((ext_vector_type(16)));
typedef float  v8f   __attribute__((ext_vector_type(8)));
typedef unsigned int v8u __attribute__((ext_vector_type(8)));

#define B_BATCH 32
#define T_SEQ   2048
#define HID     1024
#define ATTN    512
#define NGENES  2048

__device__ __forceinline__ unsigned short f2bf(float f) {
    unsigned int u = __builtin_bit_cast(unsigned int, f);
    unsigned int r = u + 0x7fffu + ((u >> 16) & 1u);   // round-to-nearest-even
    return (unsigned short)(r >> 16);
}

// ---------- A0: gc[b,g] = sum_f genes[b,g,f] * wf[f] ----------
__global__ void __launch_bounds__(256) k_collapse(const float* __restrict__ genes,
                                                  const float* __restrict__ wf,
                                                  float* __restrict__ gc, int n) {
    int idx = blockIdx.x * 256 + threadIdx.x;
    if (idx < n) {
        float4 g = ((const float4*)genes)[idx];
        gc[idx] = g.x * wf[0] + g.y * wf[1] + g.z * wf[2] + g.w * wf[3];
    }
}

// ---------- P: pack w_smiles (f32) into bf16 WMMA B-fragment layout ----------
// Fragment id = (ks*32 + jt)*32 + lane, 32 bytes each.
// lane<16:  holds col n = jt*16+lane, K = kb..kb+7 and kb+16..kb+23   (kb = 32*ks)
// lane>=16: holds col n = jt*16+(lane-16), K offsets +8 of the above.
__global__ void __launch_bounds__(256) k_packB(const float* __restrict__ w_smiles,
                                               unsigned int* __restrict__ wsB) {
    int id = blockIdx.x * 256 + threadIdx.x;          // 0 .. 32767
    int ks   = id >> 10;
    int rem  = id & 1023;
    int jt   = rem >> 5;
    int lane = rem & 31;
    int n  = jt * 16 + (lane & 15);
    int kb = ks * 32 + ((lane >> 4) << 3);
    unsigned int o[8];
#pragma unroll
    for (int q = 0; q < 8; ++q) {
        int p0 = 2 * q;
        int k0 = kb + (p0 < 8 ? p0 : 16 + (p0 - 8));
        unsigned int lo = f2bf(w_smiles[k0 * ATTN + n]);
        unsigned int hi = f2bf(w_smiles[(k0 + 1) * ATTN + n]);
        o[q] = lo | (hi << 16);
    }
    uint4* dst = (uint4*)(wsB + id * 8);
    dst[0] = make_uint4(o[0], o[1], o[2], o[3]);
    dst[1] = make_uint4(o[4], o[5], o[6], o[7]);
}

// ---------- A1: xprime[b,a] = gc[b,:]·w_genes[:,a] + b_genes[a] + b_smiles[a] ----------
__global__ void __launch_bounds__(128) k_geneproj(const float* __restrict__ gc,
                                                  const float* __restrict__ w_genes,
                                                  const float* __restrict__ b_genes,
                                                  const float* __restrict__ b_smiles,
                                                  float* __restrict__ xprime) {
    __shared__ float gs[NGENES];
    int b = blockIdx.x, tid = threadIdx.x;
    for (int i = tid; i < NGENES; i += 128) gs[i] = gc[b * NGENES + i];
    __syncthreads();
    int a = blockIdx.y * 128 + tid;
    float acc = b_genes[a] + b_smiles[a];
    for (int g = 0; g < NGENES; ++g) acc += gs[g] * w_genes[g * ATTN + a];
    xprime[b * ATTN + a] = acc;
}

// ---------- B: fused  scores[b,t] = v · tanh(xprime[b,:] + smiles[b,t,:] @ w_smiles) ----------
#define AP 1040   // padded bf16 row stride for the LDS A-tile
__global__ void __launch_bounds__(256) k_scores_wmma(const float* __restrict__ smiles,
                                                     const float* __restrict__ xprime,
                                                     const unsigned int* __restrict__ wsB,
                                                     const float* __restrict__ v,
                                                     float* __restrict__ scores) {
    __shared__ unsigned short tileA[16 * AP];   // 16 rows x 1024 bf16 (padded)
    __shared__ float part[8][16];

    int tid = threadIdx.x;
    long r0 = (long)blockIdx.x * 16;            // 16 rows of flattened [B*T]
    const float* src = smiles + r0 * HID;

    // Load 16x1024 f32 -> bf16 LDS tile (pack 2 per dword). Uniform trip count.
    for (int i = tid; i < 16 * 512; i += 256) {
        int row = i >> 9;
        int c2  = (i & 511) * 2;
        float a0 = src[row * HID + c2];
        float a1 = src[row * HID + c2 + 1];
        unsigned int p = (unsigned int)f2bf(a0) | ((unsigned int)f2bf(a1) << 16);
        *(unsigned int*)&tileA[row * AP + c2] = p;
    }
    __syncthreads();

    int wave = tid >> 5, lane = tid & 31;
    int arow  = lane & 15;
    int khalf = (lane >> 4) << 3;               // 0 or 8

    v8f acc[4];
    v8f z = {0.f, 0.f, 0.f, 0.f, 0.f, 0.f, 0.f, 0.f};
#pragma unroll
    for (int j = 0; j < 4; ++j) acc[j] = z;

    for (int ks = 0; ks < 32; ++ks) {
        int sidx = arow * AP + ks * 32 + khalf;
        const uint4* pa = (const uint4*)&tileA[sidx];
        uint4 lo = pa[0];                       // K = kb .. kb+7
        uint4 hi = pa[2];                       // K = kb+16 .. kb+23
        v8u ar;
        ar[0] = lo.x; ar[1] = lo.y; ar[2] = lo.z; ar[3] = lo.w;
        ar[4] = hi.x; ar[5] = hi.y; ar[6] = hi.z; ar[7] = hi.w;
        v16bf afrag = __builtin_bit_cast(v16bf, ar);
#pragma unroll
        for (int j = 0; j < 4; ++j) {
            int jt = wave * 4 + j;
            v8u br = ((const v8u*)wsB)[(ks * 32 + jt) * 32 + lane];
            v16bf bfrag = __builtin_bit_cast(v16bf, br);
            acc[j] = __builtin_amdgcn_wmma_f32_16x16x32_bf16(
                false, afrag, false, bfrag, (short)0, acc[j], false, false);
        }
    }

    // Fused epilogue: tanh(x + y) dotted with v, reduced over N.
    int b = (int)(r0 >> 11);
    const float* xp = xprime + b * ATTN;
    float partial[8];
#pragma unroll
    for (int i = 0; i < 8; ++i) partial[i] = 0.f;
#pragma unroll
    for (int j = 0; j < 4; ++j) {
        int n = wave * 64 + j * 16 + (lane & 15);
        float xn = xp[n];
        float vn = v[n];
#pragma unroll
        for (int i = 0; i < 8; ++i)
            partial[i] += vn * tanhf(xn + acc[j][i]);
    }
    // Sum over the 16 lanes of each half-wave (stays within halves for mask<16).
#pragma unroll
    for (int mask = 1; mask < 16; mask <<= 1)
#pragma unroll
        for (int i = 0; i < 8; ++i)
            partial[i] += __shfl_xor(partial[i], mask, 32);

    if (lane == 0) {
#pragma unroll
        for (int i = 0; i < 8; ++i) part[wave][i] = partial[i];        // rows m=0..7
    } else if (lane == 16) {
#pragma unroll
        for (int i = 0; i < 8; ++i) part[wave][8 + i] = partial[i];    // rows m=8..15
    }
    __syncthreads();
    if (tid < 16) {
        float s = 0.f;
#pragma unroll
        for (int w = 0; w < 8; ++w) s += part[w][tid];                 // fixed order: deterministic
        scores[r0 + tid] = s;
    }
}

// ---------- C1: per-batch softmax over T=2048 ----------
__global__ void __launch_bounds__(256) k_softmax(const float* __restrict__ scores,
                                                 float* __restrict__ alphas) {
    __shared__ float red[256];
    int b = blockIdx.x, tid = threadIdx.x;
    const float* s = scores + b * T_SEQ;
    float* a = alphas + b * T_SEQ;

    float m = -3.4e38f;
    for (int i = tid; i < T_SEQ; i += 256) m = fmaxf(m, s[i]);
    red[tid] = m; __syncthreads();
    for (int st = 128; st > 0; st >>= 1) {
        if (tid < st) red[tid] = fmaxf(red[tid], red[tid + st]);
        __syncthreads();
    }
    float mx = red[0]; __syncthreads();

    float sum = 0.f;
    for (int i = tid; i < T_SEQ; i += 256) {
        float e = __expf(s[i] - mx);
        a[i] = e;
        sum += e;
    }
    red[tid] = sum; __syncthreads();
    for (int st = 128; st > 0; st >>= 1) {
        if (tid < st) red[tid] += red[tid + st];
        __syncthreads();
    }
    float inv = 1.f / red[0];
    for (int i = tid; i < T_SEQ; i += 256) a[i] *= inv;
}

// ---------- C2: out[b,h] = sum_t alpha[b,t] * smiles[b,t,h] ----------
__global__ void __launch_bounds__(128) k_wsum(const float* __restrict__ smiles,
                                              const float* __restrict__ alphas,
                                              float* __restrict__ out) {
    __shared__ float al[T_SEQ];
    int b = blockIdx.x, tid = threadIdx.x;
    for (int i = tid; i < T_SEQ; i += 128) al[i] = alphas[b * T_SEQ + i];
    __syncthreads();
    int h = blockIdx.y * 128 + tid;
    const float* sm = smiles + (long)b * T_SEQ * HID + h;
    float acc = 0.f;
#pragma unroll 4
    for (int t = 0; t < T_SEQ; ++t) acc += al[t] * sm[(long)t * HID];
    out[b * HID + h] = acc;
}

extern "C" void kernel_launch(void* const* d_in, const int* in_sizes, int n_in,
                              void* d_out, int out_size, void* d_ws, size_t ws_size,
                              hipStream_t stream) {
    const float* genes    = (const float*)d_in[0];   // [32,2048,4]
    const float* smiles   = (const float*)d_in[1];   // [32,2048,1024]
    const float* wf       = (const float*)d_in[2];   // [4]
    const float* w_genes  = (const float*)d_in[3];   // [2048,512]
    const float* b_genes  = (const float*)d_in[4];   // [512]
    const float* w_smiles = (const float*)d_in[5];   // [1024,512]
    const float* b_smiles = (const float*)d_in[6];   // [512]
    const float* v        = (const float*)d_in[7];   // [512]
    float* out = (float*)d_out;                      // [32,1024]

    char* ws = (char*)d_ws;
    float* xprime        = (float*)(ws);                         //  64 KB
    float* gc            = (float*)(ws + 65536);                 // 256 KB
    float* scores        = (float*)(ws + 65536 + 262144);        // 256 KB
    float* alphas        = (float*)(ws + 65536 + 2 * 262144);    // 256 KB
    unsigned int* wsB    = (unsigned int*)(ws + 65536 + 3 * 262144); // 1 MB bf16 B-fragments

    // A0: collapse gene features (65536 outputs)
    k_collapse<<<dim3(256), dim3(256), 0, stream>>>(genes, wf, gc, B_BATCH * NGENES);
    // P: pack w_smiles into WMMA B layout (32768 fragments-lanes)
    k_packB<<<dim3(128), dim3(256), 0, stream>>>(w_smiles, wsB);
    // A1: gene projection + both biases
    k_geneproj<<<dim3(B_BATCH, ATTN / 128), dim3(128), 0, stream>>>(
        gc, w_genes, b_genes, b_smiles, xprime);
    // B: fused WMMA projection + tanh + dot(v) -> scores  (4096 tiles of 16 rows)
    k_scores_wmma<<<dim3((B_BATCH * T_SEQ) / 16), dim3(256), 0, stream>>>(
        smiles, xprime, wsB, v, scores);
    // C1: softmax per batch
    k_softmax<<<dim3(B_BATCH), dim3(256), 0, stream>>>(scores, alphas);
    // C2: attention-weighted sum (second and final pass over smiles)
    k_wsum<<<dim3(B_BATCH, HID / 128), dim3(128), 0, stream>>>(smiles, alphas, out);
}
